// LaseBlock_42571715838401
// MI455X (gfx1250) — matrix operand order repaired and checked by reading, not compiled, not run on hardware
//
#include <hip/hip_runtime.h>
#include <hip/hip_fp16.h>

typedef __attribute__((ext_vector_type(16))) _Float16 v16h;
typedef __attribute__((ext_vector_type(8)))  _Float16 v8h;
typedef __attribute__((ext_vector_type(8)))  float    v8f;

static constexpr int Nn = 8192;
static constexpr int Dd = 128;
static constexpr int E1c = 262144, E2c = 262144;

// ---- workspace layout (bytes) ----
static constexpr size_t SZ_BITSET = (size_t)Nn * Nn / 8;            // 8 MiB
static constexpr size_t OFF_AM  = 0;
static constexpr size_t OFF_A1S = OFF_AM  + SZ_BITSET;
static constexpr size_t OFF_A2S = OFF_A1S + SZ_BITSET;
static constexpr size_t OFF_Y   = OFF_A2S + SZ_BITSET;              // f32 N*D
static constexpr size_t OFF_X2  = OFF_Y   + (size_t)Nn * Dd * 4;    // f32 N*D
static constexpr size_t OFF_CNT = OFF_X2  + (size_t)Nn * Dd * 4;    // 2 ints
static constexpr size_t ZERO_BYTES = OFF_CNT + 256;                 // zero every call
static constexpr size_t OFF_L1  = ZERO_BYTES;
static constexpr size_t OFF_L2  = OFF_L1 + (size_t)2 * E1c * 4;
static constexpr size_t OFF_XH  = OFF_L2 + (size_t)2 * E2c * 4;     // f16 x
static constexpr size_t OFF_YH  = OFF_XH + (size_t)Nn * Dd * 2;     // f16 y
static constexpr size_t OFF_G0  = OFF_YH + (size_t)Nn * Dd * 2;
static constexpr size_t OFF_G1  = OFF_G0 + (size_t)Nn * Dd * 4;
static constexpr size_t OFF_Q   = OFF_G1 + (size_t)Nn * Dd * 4;
static constexpr size_t OFF_K   = OFF_Q  + (size_t)Nn * Dd * 4;
static constexpr size_t OFF_V   = OFF_K  + (size_t)Nn * Dd * 4;
static constexpr size_t OFF_WH  = OFF_V  + (size_t)Nn * Dd * 4;     // 5x f16 128x128
// total ~60.2 MiB

// ------------------------------------------------------------------
__global__ void k_zero(uint4* p, long n16) {
  long i = (long)blockIdx.x * blockDim.x + threadIdx.x;
  long stride = (long)gridDim.x * blockDim.x;
  uint4 z = {0u, 0u, 0u, 0u};
  for (; i < n16; i += stride) p[i] = z;
}

__global__ void k_scatter_mask(const int* __restrict__ row,
                               const int* __restrict__ col,
                               int E, unsigned* __restrict__ am) {
  int i = blockIdx.x * blockDim.x + threadIdx.x;
  int stride = gridDim.x * blockDim.x;
  for (; i < E; i += stride) {
    unsigned idx = (unsigned)row[i] * (unsigned)Nn + (unsigned)col[i];
    atomicOr(&am[idx >> 5], 1u << (idx & 31u));
  }
}

// Keep edges that survive the mask; exact dedup via test-and-set bitset
// (matches ((adj * mask) > 0) binary semantics).
__global__ void k_filter(const int* __restrict__ src, const int* __restrict__ dst,
                         int E, const unsigned* __restrict__ am,
                         unsigned* __restrict__ aset,
                         int* __restrict__ cnt, int* __restrict__ list) {
  int i = blockIdx.x * blockDim.x + threadIdx.x;
  int stride = gridDim.x * blockDim.x;
  for (; i < E; i += stride) {
    int s = src[i], d = dst[i];
    unsigned idx = (unsigned)s * (unsigned)Nn + (unsigned)d;
    unsigned bit = 1u << (idx & 31u);
    if (am[idx >> 5] & bit) {
      unsigned old = atomicOr(&aset[idx >> 5], bit);
      if (!(old & bit)) {
        int p = atomicAdd(cnt, 1);
        list[2 * p] = s;
        list[2 * p + 1] = d;
      }
    }
  }
}

__global__ void k_f32_to_f16(const float* __restrict__ src,
                             _Float16* __restrict__ dst, int n) {
  int i = blockIdx.x * blockDim.x + threadIdx.x;
  int stride = gridDim.x * blockDim.x;
  for (; i < n; i += stride) dst[i] = (_Float16)src[i];
}

// y[d,:] += x[s,:]  for each surviving edge (s->d) of A1   (== A1.T @ x)
__global__ void k_spmm_edges(const int* __restrict__ cnt, const int* __restrict__ list,
                             const float* __restrict__ x, float* __restrict__ y) {
  int wave = (blockIdx.x * blockDim.x + threadIdx.x) >> 5;
  int lane = threadIdx.x & 31;
  int nwaves = (gridDim.x * blockDim.x) >> 5;
  int m = *cnt;
  for (int e = wave; e < m; e += nwaves) {
    int s = list[2 * e], d = list[2 * e + 1];
    #pragma unroll
    for (int t = 0; t < Dd; t += 32)
      atomicAdd(&y[(size_t)d * Dd + t + lane], x[(size_t)s * Dd + t + lane]);
  }
}

// x2[d,:] += (q_d . k_s) * v[s,:]  for each surviving edge (s->d) of A2
__global__ void k_attn_edges(const int* __restrict__ cnt, const int* __restrict__ list,
                             const float* __restrict__ q, const float* __restrict__ k,
                             const float* __restrict__ v, float* __restrict__ x2) {
  int wave = (blockIdx.x * blockDim.x + threadIdx.x) >> 5;
  int lane = threadIdx.x & 31;
  int nwaves = (gridDim.x * blockDim.x) >> 5;
  int m = *cnt;
  for (int e = wave; e < m; e += nwaves) {
    int s = list[2 * e], d = list[2 * e + 1];
    float acc = 0.f;
    #pragma unroll
    for (int t = 0; t < Dd; t += 32)
      acc += q[(size_t)d * Dd + t + lane] * k[(size_t)s * Dd + t + lane];
    #pragma unroll
    for (int off = 16; off > 0; off >>= 1) acc += __shfl_xor(acc, off, 32);
    #pragma unroll
    for (int t = 0; t < Dd; t += 32)
      atomicAdd(&x2[(size_t)d * Dd + t + lane], acc * v[(size_t)s * Dd + t + lane]);
  }
}

// C[M x 128] = A[M x 128] @ W^T (W is 128x128 row-major), f16 in / f32 out.
// One wave32 computes one 16x16 tile via 4x v_wmma_f32_16x16x32_f16.
__global__ __launch_bounds__(128)
void k_gemm_xwT(const _Float16* __restrict__ A, const _Float16* __restrict__ W,
                float* __restrict__ C, int M) {
  const int lane   = threadIdx.x & 31;
  const int waveId = threadIdx.x >> 5;
  const int colTiles = Dd / 16;                       // 8
  int tile = blockIdx.x * 4 + waveId;
  int rowTile = tile / colTiles;
  int colTile = tile - rowTile * colTiles;
  if (rowTile * 16 >= M) return;

  const int hi  = lane >> 4;                          // lane half
  const int row = rowTile * 16 + (lane & 15);         // A: lane holds row M
  const int col = colTile * 16 + (lane & 15);         // B: lane holds col N

  v8f acc = {};
  #pragma unroll
  for (int k0 = 0; k0 < Dd; k0 += 32) {
    // A frag (16x32 f16): halves 0..7 -> K = k0+16*hi .. +7 ; halves 8..15 -> K += 8
    v8h a_lo = *(const v8h*)(A + (size_t)row * Dd + k0 + 16 * hi);
    v8h a_hi = *(const v8h*)(A + (size_t)row * Dd + k0 + 16 * hi + 8);
    v16h af;
    #pragma unroll
    for (int i = 0; i < 8; ++i) { af[i] = a_lo[i]; af[8 + i] = a_hi[i]; }
    // B frag (32x16 f16): B[k][n] = W[n][k]; lanes 0-15 hold K=k0..k0+15,
    // lanes 16-31 hold K=k0+16..k0+31 (contiguous k order, same as A)
    v8h b0 = *(const v8h*)(W + (size_t)col * Dd + k0 + 16 * hi);
    v8h b1 = *(const v8h*)(W + (size_t)col * Dd + k0 + 16 * hi + 8);
    v16h bf;
    #pragma unroll
    for (int i = 0; i < 8; ++i) { bf[i] = b0[i]; bf[8 + i] = b1[i]; }
    acc = __builtin_amdgcn_wmma_f32_16x16x32_f16(false, af, false, bf,
                                                 (short)0, acc, false, false);
  }
  // C/D layout: VGPR r -> M = rowTile*16 + r + 8*hi, N = colTile*16 + (lane&15)
  int n = colTile * 16 + (lane & 15);
  int mbase = rowTile * 16 + 8 * hi;
  #pragma unroll
  for (int r = 0; r < 8; ++r) C[(size_t)(mbase + r) * Dd + n] = acc[r];
}

// out = (g0 + g1)/np1 + (np1-1)/np1 * x - x2/np2 ,  np2 = |A2| / N (device)
__global__ void k_combine(const float* __restrict__ x, const float* __restrict__ g0,
                          const float* __restrict__ g1, const float* __restrict__ x2,
                          const int* __restrict__ cnt2, float np1,
                          float* __restrict__ out, int n) {
  float inv1 = 1.0f / np1;
  float c1 = (np1 - 1.0f) * inv1;
  float np2 = (float)(*cnt2) / (float)Nn;
  float inv2 = 1.0f / np2;
  int i = blockIdx.x * blockDim.x + threadIdx.x;
  int stride = gridDim.x * blockDim.x;
  for (; i < n; i += stride)
    out[i] = (g0[i] + g1[i]) * inv1 + c1 * x[i] - x2[i] * inv2;
}

// ------------------------------------------------------------------
extern "C" void kernel_launch(void* const* d_in, const int* in_sizes, int n_in,
                              void* d_out, int out_size, void* d_ws, size_t ws_size,
                              hipStream_t stream) {
  const float* x   = (const float*)d_in[0];
  const int*   e1  = (const int*)d_in[1];
  const int*   e2  = (const int*)d_in[2];
  const int*   mk  = (const int*)d_in[3];
  const float* Wd[5] = { (const float*)d_in[4], (const float*)d_in[5],
                         (const float*)d_in[6], (const float*)d_in[7],
                         (const float*)d_in[8] };
  const int E1 = in_sizes[1] / 2;
  const int E2 = in_sizes[2] / 2;
  const int EM = in_sizes[3] / 2;
  const float np1 = (float)EM / (float)Nn;   // N * (EM / N^2)

  char* ws = (char*)d_ws;
  unsigned* AM  = (unsigned*)(ws + OFF_AM);
  unsigned* A1S = (unsigned*)(ws + OFF_A1S);
  unsigned* A2S = (unsigned*)(ws + OFF_A2S);
  float*    Yb  = (float*)(ws + OFF_Y);
  float*    X2b = (float*)(ws + OFF_X2);
  int*      CNT = (int*)(ws + OFF_CNT);      // [0]=|A1| edges, [1]=|A2| edges
  int*      L1  = (int*)(ws + OFF_L1);
  int*      L2  = (int*)(ws + OFF_L2);
  _Float16* XH  = (_Float16*)(ws + OFF_XH);
  _Float16* YH  = (_Float16*)(ws + OFF_YH);
  float*    G0  = (float*)(ws + OFF_G0);
  float*    G1  = (float*)(ws + OFF_G1);
  float*    Qb  = (float*)(ws + OFF_Q);
  float*    Kb  = (float*)(ws + OFF_K);
  float*    Vb  = (float*)(ws + OFF_V);
  _Float16* WH  = (_Float16*)(ws + OFF_WH);

  // 1) zero bitsets + accumulators + counters (every call: deterministic)
  k_zero<<<2048, 256, 0, stream>>>((uint4*)ws, (long)(ZERO_BYTES / 16));
  // 2) mask adjacency bitset
  k_scatter_mask<<<2048, 256, 0, stream>>>(mk, mk + EM, EM, AM);
  // 3) filter + dedup the two edge lists against the mask
  k_filter<<<1024, 256, 0, stream>>>(e1, e1 + E1, E1, AM, A1S, CNT + 0, L1);
  k_filter<<<1024, 256, 0, stream>>>(e2, e2 + E2, E2, AM, A2S, CNT + 1, L2);
  // 4) f16 copies for WMMA
  k_f32_to_f16<<<1024, 256, 0, stream>>>(x, XH, Nn * Dd);
  for (int i = 0; i < 5; ++i)
    k_f32_to_f16<<<64, 256, 0, stream>>>(Wd[i], WH + (size_t)i * Dd * Dd, Dd * Dd);
  // 5) y = A1.T @ x via sparse scatter (~2K edges), then to f16
  k_spmm_edges<<<1024, 256, 0, stream>>>(CNT + 0, L1, x, Yb);
  k_f32_to_f16<<<1024, 256, 0, stream>>>(Yb, YH, Nn * Dd);
  // 6) five WMMA GEMMs: 8192x128 @ 128x128^T, 4096 tiles / 4 waves per block
  const int gemmBlocks = (Nn / 16) * (Dd / 16) / 4;   // 1024
  k_gemm_xwT<<<gemmBlocks, 128, 0, stream>>>(XH, WH + 0 * Dd * Dd, G0, Nn);
  k_gemm_xwT<<<gemmBlocks, 128, 0, stream>>>(YH, WH + 1 * Dd * Dd, G1, Nn);
  k_gemm_xwT<<<gemmBlocks, 128, 0, stream>>>(XH, WH + 2 * Dd * Dd, Qb, Nn);
  k_gemm_xwT<<<gemmBlocks, 128, 0, stream>>>(XH, WH + 3 * Dd * Dd, Kb, Nn);
  k_gemm_xwT<<<gemmBlocks, 128, 0, stream>>>(XH, WH + 4 * Dd * Dd, Vb, Nn);
  // 7) masked unnormalized attention as sparse edge scatter
  k_attn_edges<<<1024, 256, 0, stream>>>(CNT + 1, L2, Qb, Kb, Vb, X2b);
  // 8) epilogue: out = (g0+g1)/np1 + (np1-1)/np1*x - x2/np2
  k_combine<<<2048, 256, 0, stream>>>(x, G0, G1, X2b, CNT + 1, np1,
                                      (float*)d_out, Nn * Dd);
}
